// AudioToJoints_34196529610958
// MI455X (gfx1250) — matrix operand order repaired and checked by reading, not compiled, not run on hardware
//
#include <hip/hip_runtime.h>
#include <math.h>

// ---------------------------------------------------------------------------
// Persistent bf16-WMMA LSTM for MI455X (gfx1250, wave32).
//
//   B=32, T=2048, I=128, H=1024, O=51.
//   32 persistent workgroups; WG g owns hidden units [g*32, g*32+32)
//   => 128 gate rows (i,f,g,o x 32) = 8 WMMA N-tiles. 8 waves/WG, wave w owns
//   N-tile w for both M-tiles (batches 0-15, 16-31).
//   Per step: gates = x_t @ W_ih_sliceT + h @ W_hh_sliceT  via
//   v_wmma_f32_16x16x32_bf16, K = 128 (input) + 1024 (recurrent).
//   h is stored in global memory directly in WMMA A-fragment layout so the
//   next step (and the final FC) load operands contiguously.
//   Grid-wide sync per step: atomic counter + s_sleep spin.
// ---------------------------------------------------------------------------

typedef __attribute__((ext_vector_type(16))) __bf16 v16bf;
typedef __attribute__((ext_vector_type(8)))  float  v8f;

#define B_   32
#define T_   2048
#define I_   128
#define H_   1024
#define O_   51
#define NWG  32            // persistent workgroups
#define HJ   32            // hidden units per WG
#define FRAG 512           // elements per 16x32 / 32x16 bf16 fragment

// Workspace byte offsets (all 256-aligned)
#define OFF_CNT   ((size_t)0)
#define OFF_BIAS  ((size_t)256)                        // 4096 f32
#define OFF_WIH   ((size_t)16640)                      // 32*8*4*512 bf16 = 1 MB
#define OFF_WHH   ((size_t)1065216)                    // 32*8*32*512 bf16 = 8 MB
#define OFF_WFC   ((size_t)9453824)                    // 4*32*512 bf16 = 128 KB
#define OFF_XF    ((size_t)9584896)                    // 2048*2*4*512 bf16 = 16 MB
#define OFF_HSF   ((size_t)26362112)                   // 2049*2*32*512 bf16 ~ 134 MB

// 16-bit A-matrix 16x32 fragment index (ISA 7.12.2): lane = M + 16*half,
// half=(K>>3)&1, elem = (K&7) + 8*(K>>4).
__device__ __forceinline__ int a_idx(int M, int K) {
  int half = (K >> 3) & 1;
  int e    = (K & 7) + ((K >> 4) << 3);
  return (M + (half << 4)) * 16 + e;
}

// ------------------------------- pack kernels ------------------------------

__global__ void k_pack_x(const float* __restrict__ x, __bf16* __restrict__ xf) {
  int idx = blockIdx.x * blockDim.x + threadIdx.x;      // B*T*I
  if (idx >= B_ * T_ * I_) return;
  int i = idx % I_;
  int t = (idx / I_) % T_;
  int b = idx / (I_ * T_);
  int m = b >> 4, M = b & 15, kb = i >> 5, K = i & 31;
  xf[(((size_t)t * 2 + m) * 4 + kb) * FRAG + a_idx(M, K)] = (__bf16)x[idx];
}

__global__ void k_pack_wih(const float* __restrict__ wih, __bf16* __restrict__ dst) {
  int idx = blockIdx.x * blockDim.x + threadIdx.x;      // 32*8*4*512
  if (idx >= NWG * 8 * 4 * FRAG) return;
  int n  = (idx >> 4) & 15;
  int k  = (idx & 7) + (((idx >> 8) & 1) << 3) + (((idx >> 3) & 1) << 4);
  int kb = (idx >> 9) & 3;
  int w  = (idx >> 11) & 7;
  int g  = idx >> 14;
  int rl = w * 16 + n;                 // local gate row 0..127
  int grow = (rl >> 5) * H_ + g * HJ + (rl & 31);
  dst[idx] = (__bf16)wih[grow * I_ + kb * 32 + k];      // B-frag: lane=N+16*half, elem=K (table order)
}

__global__ void k_pack_whh(const float* __restrict__ whh, __bf16* __restrict__ dst) {
  int idx = blockIdx.x * blockDim.x + threadIdx.x;      // 32*8*32*512
  if (idx >= NWG * 8 * 32 * FRAG) return;
  int n  = (idx >> 4) & 15;
  int k  = (idx & 7) + (((idx >> 8) & 1) << 3) + (((idx >> 3) & 1) << 4);
  int kb = (idx >> 9) & 31;
  int w  = (idx >> 14) & 7;
  int g  = idx >> 17;
  int rl = w * 16 + n;
  int grow = (rl >> 5) * H_ + g * HJ + (rl & 31);
  dst[idx] = (__bf16)whh[grow * H_ + kb * 32 + k];
}

__global__ void k_pack_wfc(const float* __restrict__ wfc, __bf16* __restrict__ dst) {
  int idx = blockIdx.x * blockDim.x + threadIdx.x;      // 4*32*512
  if (idx >= 4 * 32 * FRAG) return;
  int n  = (idx >> 4) & 15;
  int k  = (idx & 7) + (((idx >> 8) & 1) << 3) + (((idx >> 3) & 1) << 4);
  int kb = (idx >> 9) & 31;
  int nt = idx >> 14;
  int o  = nt * 16 + n;
  dst[idx] = (o < O_) ? (__bf16)wfc[o * H_ + kb * 32 + k] : (__bf16)0.f;
}

__global__ void k_pack_misc(const float* __restrict__ b_ih, const float* __restrict__ b_hh,
                            float* __restrict__ bias, __bf16* __restrict__ hsf0,
                            int* __restrict__ cnt) {
  int idx = blockIdx.x * blockDim.x + threadIdx.x;      // 2*32*512 = 32768
  if (idx >= 2 * 32 * FRAG) return;
  hsf0[idx] = (__bf16)0.f;                              // h_{-1} = 0
  if (idx < 4 * H_) bias[idx] = b_ih[idx] + b_hh[idx];
  if (idx == 0)     *cnt = 0;
}

// ---------------------------- persistent LSTM ------------------------------

__global__ __launch_bounds__(256)
void k_lstm(const __bf16* __restrict__ xf,  const __bf16* __restrict__ wihf,
            const __bf16* __restrict__ whhf, const float* __restrict__ bias,
            __bf16* __restrict__ hsf, int* __restrict__ cnt) {
  __shared__ float s_gates[32][128];   // [batch][local gate row]
  __shared__ float s_c[32][HJ];        // persistent cell state slice

  const int tid = threadIdx.x;
  const int wg  = blockIdx.x;          // 0..31
  const int w   = tid >> 5;            // wave 0..7 = N-tile
  const int ln  = tid & 31;

  for (int u = tid; u < 32 * HJ; u += 256) (&s_c[0][0])[u] = 0.f;
  __syncthreads();

  volatile int* vcnt = cnt;
  const v16bf* BwI = (const v16bf*)(wihf + ((size_t)(wg * 8 + w) * 4)  * FRAG) + ln;
  const v16bf* BwH = (const v16bf*)(whhf + ((size_t)(wg * 8 + w) * 32) * FRAG) + ln;

  for (int t = 0; t < T_; ++t) {
    v8f acc0 = {};
    v8f acc1 = {};

    // -------- input contribution: K = 128 --------
    {
      const v16bf* A0 = (const v16bf*)(xf + ((size_t)t * 2 + 0) * 4 * FRAG) + ln;
      const v16bf* A1 = (const v16bf*)(xf + ((size_t)t * 2 + 1) * 4 * FRAG) + ln;
#pragma unroll
      for (int kb = 0; kb < 4; ++kb) {
        v16bf a0 = A0[kb * 32];
        v16bf a1 = A1[kb * 32];
        v16bf bb = BwI[kb * 32];
        acc0 = __builtin_amdgcn_wmma_f32_16x16x32_bf16(false, a0, false, bb, (short)0, acc0, false, false);
        acc1 = __builtin_amdgcn_wmma_f32_16x16x32_bf16(false, a1, false, bb, (short)0, acc1, false, false);
      }
    }

    // -------- recurrent contribution: K = 1024 --------
    {
      const v16bf* A0 = (const v16bf*)(hsf + (((size_t)t * 2 + 0) * 32) * FRAG) + ln;
      const v16bf* A1 = (const v16bf*)(hsf + (((size_t)t * 2 + 1) * 32) * FRAG) + ln;
#pragma unroll 4
      for (int kb = 0; kb < 32; ++kb) {
        __builtin_prefetch((const void*)(BwH + (kb + 2) * 32), 0, 1);  // global_prefetch
        v16bf a0 = A0[kb * 32];
        v16bf a1 = A1[kb * 32];
        v16bf bb = BwH[kb * 32];
        acc0 = __builtin_amdgcn_wmma_f32_16x16x32_bf16(false, a0, false, bb, (short)0, acc0, false, false);
        acc1 = __builtin_amdgcn_wmma_f32_16x16x32_bf16(false, a1, false, bb, (short)0, acc1, false, false);
      }
    }

    // -------- spill gate tiles to LDS (C/D layout: lane=N+16*(M/8), vgpr=M%8)
    {
      int Mh = (ln >> 4) << 3;          // 0 or 8
      int N  = ln & 15;
#pragma unroll
      for (int v = 0; v < 8; ++v) {
        s_gates[Mh + v][w * 16 + N]      = acc0[v];
        s_gates[16 + Mh + v][w * 16 + N] = acc1[v];
      }
    }
    __syncthreads();

    // -------- fused gate nonlinearity + cell update (4 units / thread) -----
#pragma unroll
    for (int u = 0; u < 4; ++u) {
      int idx = tid * 4 + u;            // 0..1023 = batch*32 + unit
      int b   = idx >> 5;
      int jl  = idx & 31;
      int jg  = wg * HJ + jl;
      float gi = s_gates[b][jl]      + bias[jg];
      float gf = s_gates[b][32 + jl] + bias[H_ + jg];
      float gg = s_gates[b][64 + jl] + bias[2 * H_ + jg];
      float go = s_gates[b][96 + jl] + bias[3 * H_ + jg];
      float ig = 1.f / (1.f + __expf(-gi));
      float fg = 1.f / (1.f + __expf(-gf));
      float gt = tanhf(gg);
      float og = 1.f / (1.f + __expf(-go));
      float c  = fg * s_c[b][jl] + ig * gt;
      s_c[b][jl] = c;
      float h  = og * tanhf(c);
      // scatter h into A-fragment layout of slot t+1, K-block == wg
      int m = b >> 4, M = b & 15;
      hsf[(((size_t)(t + 1) * 2 + m) * 32 + wg) * FRAG + a_idx(M, jl)] = (__bf16)h;
    }

    // -------- grid-wide barrier between timesteps --------
    __threadfence();
    __syncthreads();
    if (tid == 0) {
      atomicAdd(cnt, 1);
      int target = NWG * (t + 1);
      while (vcnt[0] < target) __builtin_amdgcn_s_sleep(2);
    }
    __syncthreads();
    __threadfence();
  }
}

// ------------------------------- final FC ----------------------------------

__global__ __launch_bounds__(256)
void k_fc(const __bf16* __restrict__ hsf, const __bf16* __restrict__ wfcf,
          const float* __restrict__ bfc, float* __restrict__ out) {
  int tid = threadIdx.x;
  int w   = tid >> 5, ln = tid & 31;
  int gw  = blockIdx.x * 8 + w;         // 0..16383
  int nt  = gw & 3;                     // output N-tile (O padded to 64)
  int mt  = gw >> 2;                    // 0..4095 = t*2 + m
  int t   = mt >> 1;
  int m   = mt & 1;

  const v16bf* A  = (const v16bf*)(hsf + (((size_t)(t + 1) * 2 + m) * 32) * FRAG) + ln;
  const v16bf* Bw = (const v16bf*)(wfcf + (size_t)nt * 32 * FRAG) + ln;
  v8f acc = {};
#pragma unroll 4
  for (int kb = 0; kb < 32; ++kb) {
    v16bf a  = A[kb * 32];
    v16bf bb = Bw[kb * 32];
    acc = __builtin_amdgcn_wmma_f32_16x16x32_bf16(false, a, false, bb, (short)0, acc, false, false);
  }
  int Mh = (ln >> 4) << 3;
  int N  = ln & 15;
  int o  = nt * 16 + N;
  if (o < O_) {
    float bo = bfc[o];
#pragma unroll
    for (int v = 0; v < 8; ++v) {
      int b = m * 16 + Mh + v;
      out[((size_t)b * T_ + t) * O_ + o] = acc[v] + bo;
    }
  }
}

// ------------------------------- launcher ----------------------------------

extern "C" void kernel_launch(void* const* d_in, const int* in_sizes, int n_in,
                              void* d_out, int out_size, void* d_ws, size_t ws_size,
                              hipStream_t stream) {
  const float* x    = (const float*)d_in[0];
  const float* W_ih = (const float*)d_in[1];
  const float* W_hh = (const float*)d_in[2];
  const float* b_ih = (const float*)d_in[3];
  const float* b_hh = (const float*)d_in[4];
  const float* W_fc = (const float*)d_in[5];
  const float* b_fc = (const float*)d_in[6];
  float* out = (float*)d_out;

  char* ws = (char*)d_ws;
  int*    cnt  = (int*)   (ws + OFF_CNT);
  float*  bias = (float*) (ws + OFF_BIAS);
  __bf16* wihf = (__bf16*)(ws + OFF_WIH);
  __bf16* whhf = (__bf16*)(ws + OFF_WHH);
  __bf16* wfcf = (__bf16*)(ws + OFF_WFC);
  __bf16* xf   = (__bf16*)(ws + OFF_XF);
  __bf16* hsf  = (__bf16*)(ws + OFF_HSF);

  k_pack_x  <<<(B_ * T_ * I_ + 255) / 256, 256, 0, stream>>>(x, xf);
  k_pack_wih<<<(NWG * 8 * 4  * FRAG) / 256, 256, 0, stream>>>(W_ih, wihf);
  k_pack_whh<<<(NWG * 8 * 32 * FRAG) / 256, 256, 0, stream>>>(W_hh, whhf);
  k_pack_wfc<<<(4 * 32 * FRAG) / 256, 256, 0, stream>>>(W_fc, wfcf);
  k_pack_misc<<<(2 * 32 * FRAG) / 256, 256, 0, stream>>>(b_ih, b_hh, bias, hsf, cnt);

  k_lstm<<<NWG, 256, 0, stream>>>(xf, wihf, whhf, bias, hsf, cnt);
  k_fc  <<<(T_ * 2 * 4) / 8, 256, 0, stream>>>(hsf, wfcf, b_fc, out);
}
